// MyGCN_51410758533499
// MI455X (gfx1250) — matrix operand order repaired and checked by reading, not compiled, not run on hardware
//
#include <hip/hip_runtime.h>
#include <hip/hip_bf16.h>

#define N_NODES   100000
#define N_EDGES   3200000
#define N_FEAT    256
#define HIDDEN    64
#define N_CLASSES 40
#define NPAD2     48          // N_CLASSES padded to 3 WMMA tiles

typedef __attribute__((ext_vector_type(16))) __bf16 v16bf;
typedef __attribute__((ext_vector_type(8)))  float  v8f;

// ---------------------------------------------------------------- utilities
__global__ void gcn_fill0(float* __restrict__ p, int n) {
    int i = blockIdx.x * blockDim.x + threadIdx.x;
    if (i < n) p[i] = 0.0f;
}

// deg1[d] += ew[e]; deg2[d] += 1
__global__ void gcn_degree(const int* __restrict__ ei, const float* __restrict__ ew,
                           float* __restrict__ deg1, float* __restrict__ deg2) {
    int e = blockIdx.x * blockDim.x + threadIdx.x;
    if (e >= N_EDGES) return;
    int d = ei[N_EDGES + e];
    atomicAdd(&deg1[d], ew[e]);
    atomicAdd(&deg2[d], 1.0f);
}

// in-place deg -> rsqrt(deg + 1)
__global__ void gcn_rsqrt(float* __restrict__ d1, float* __restrict__ d2) {
    int i = blockIdx.x * blockDim.x + threadIdx.x;
    if (i >= N_NODES) return;
    d1[i] = rsqrtf(d1[i] + 1.0f);
    d2[i] = rsqrtf(d2[i] + 1.0f);
}

// WT1[n][k] = bf16(W1[k][n]),  n in [0,64), k in [0,256)
__global__ void gcn_prepW1(const float* __restrict__ W, __bf16* __restrict__ WT) {
    int t = blockIdx.x * blockDim.x + threadIdx.x;
    if (t >= HIDDEN * N_FEAT) return;
    int n = t >> 8, k = t & 255;
    WT[t] = (__bf16)W[k * HIDDEN + n];
}

// WT2[n][k] = bf16(W2[k][n]) for n<40 else 0,  n in [0,48), k in [0,64)
__global__ void gcn_prepW2(const float* __restrict__ W, __bf16* __restrict__ WT) {
    int t = blockIdx.x * blockDim.x + threadIdx.x;
    if (t >= NPAD2 * HIDDEN) return;
    int n = t >> 6, k = t & 63;
    WT[t] = (n < N_CLASSES) ? (__bf16)W[k * N_CLASSES + n] : (__bf16)0.0f;
}

// ------------------------------------------------------------ WMMA GEMM #1
// h[100000,64] = x[100000,256] @ W1[256,64], B pre-transposed bf16: WT[n][k]
// One wave computes a 16x64 strip: 4 WMMA 16x16 tiles, 8 K-steps of 32.
__global__ void __launch_bounds__(128) gcn_gemm1(const float* __restrict__ x,
                                                 const __bf16* __restrict__ WT,
                                                 float* __restrict__ h) {
    const int wave  = threadIdx.x >> 5;
    const int lane  = threadIdx.x & 31;
    const int row0  = (blockIdx.x * 4 + wave) * 16;
    if (row0 >= N_NODES) return;            // uniform per wave -> EXEC stays all-1s
    const int l15   = lane & 15;
    const int khalf = lane >> 4;

    v8f acc[4] = {};
    const float* __restrict__ arow = x + (size_t)(row0 + l15) * N_FEAT;

    for (int kb = 0; kb < N_FEAT; kb += 32) {
        // A 16x32 bf16 frag: lanes<16 hold K 0..15, lanes>=16 hold 16..31
        const int ka0 = kb + khalf * 16;
        const int ka1 = kb + khalf * 16 + 8;
        const float4 f0 = *(const float4*)(arow + ka0);
        const float4 f1 = *(const float4*)(arow + ka0 + 4);
        const float4 f2 = *(const float4*)(arow + ka1);
        const float4 f3 = *(const float4*)(arow + ka1 + 4);
        v16bf a;
        a[0]=(__bf16)f0.x;  a[1]=(__bf16)f0.y;  a[2]=(__bf16)f0.z;  a[3]=(__bf16)f0.w;
        a[4]=(__bf16)f1.x;  a[5]=(__bf16)f1.y;  a[6]=(__bf16)f1.z;  a[7]=(__bf16)f1.w;
        a[8]=(__bf16)f2.x;  a[9]=(__bf16)f2.y;  a[10]=(__bf16)f2.z; a[11]=(__bf16)f2.w;
        a[12]=(__bf16)f3.x; a[13]=(__bf16)f3.y; a[14]=(__bf16)f3.z; a[15]=(__bf16)f3.w;

        // B 32x16 frags: lanes<16 hold K kb..kb+15 of column n, lanes>=16 hold kb+16..kb+31
        const int kbB = kb + khalf * 16;
#pragma unroll
        for (int nt = 0; nt < 4; ++nt) {
            const v16bf b = *(const v16bf*)(WT + (size_t)(nt * 16 + l15) * N_FEAT + kbB);
            acc[nt] = __builtin_amdgcn_wmma_f32_16x16x32_bf16(
                false, a, false, b, (short)0, acc[nt], false, false);
        }
    }
    // C/D layout: VGPR r -> row (khalf*8 + r), col l15 within tile
    const int mstore = row0 + khalf * 8;
#pragma unroll
    for (int nt = 0; nt < 4; ++nt)
#pragma unroll
        for (int r = 0; r < 8; ++r)
            h[(size_t)(mstore + r) * HIDDEN + nt * 16 + l15] = acc[nt][r];
}

// ------------------------------------------------------------ WMMA GEMM #2
// h2[100000,40] = out1[100000,64] @ W2[64,40], B pre-transposed/padded: WT[48][64]
__global__ void __launch_bounds__(128) gcn_gemm2(const float* __restrict__ x,
                                                 const __bf16* __restrict__ WT,
                                                 float* __restrict__ h) {
    const int wave  = threadIdx.x >> 5;
    const int lane  = threadIdx.x & 31;
    const int row0  = (blockIdx.x * 4 + wave) * 16;
    if (row0 >= N_NODES) return;
    const int l15   = lane & 15;
    const int khalf = lane >> 4;

    v8f acc[3] = {};
    const float* __restrict__ arow = x + (size_t)(row0 + l15) * HIDDEN;

    for (int kb = 0; kb < HIDDEN; kb += 32) {
        const int ka0 = kb + khalf * 16;
        const int ka1 = kb + khalf * 16 + 8;
        const float4 f0 = *(const float4*)(arow + ka0);
        const float4 f1 = *(const float4*)(arow + ka0 + 4);
        const float4 f2 = *(const float4*)(arow + ka1);
        const float4 f3 = *(const float4*)(arow + ka1 + 4);
        v16bf a;
        a[0]=(__bf16)f0.x;  a[1]=(__bf16)f0.y;  a[2]=(__bf16)f0.z;  a[3]=(__bf16)f0.w;
        a[4]=(__bf16)f1.x;  a[5]=(__bf16)f1.y;  a[6]=(__bf16)f1.z;  a[7]=(__bf16)f1.w;
        a[8]=(__bf16)f2.x;  a[9]=(__bf16)f2.y;  a[10]=(__bf16)f2.z; a[11]=(__bf16)f2.w;
        a[12]=(__bf16)f3.x; a[13]=(__bf16)f3.y; a[14]=(__bf16)f3.z; a[15]=(__bf16)f3.w;

        const int kbB = kb + khalf * 16;
#pragma unroll
        for (int nt = 0; nt < 3; ++nt) {
            const v16bf b = *(const v16bf*)(WT + (size_t)(nt * 16 + l15) * HIDDEN + kbB);
            acc[nt] = __builtin_amdgcn_wmma_f32_16x16x32_bf16(
                false, a, false, b, (short)0, acc[nt], false, false);
        }
    }
    const int mstore = row0 + khalf * 8;
#pragma unroll
    for (int nt = 0; nt < 3; ++nt) {
        const int n = nt * 16 + l15;
        if (n < N_CLASSES)
#pragma unroll
            for (int r = 0; r < 8; ++r)
                h[(size_t)(mstore + r) * N_CLASSES + n] = acc[nt][r];
    }
}

// ------------------------------------------------------------ edge scatter
// layer 1: 16 threads per edge, each handles 4 of 64 features (float4 reads)
__global__ void gcn_scatter1(const int* __restrict__ ei, const float* __restrict__ ew,
                             const float* __restrict__ dinv, const float* __restrict__ h,
                             float* __restrict__ agg) {
    int t = blockIdx.x * blockDim.x + threadIdx.x;
    int e = t >> 4;
    int j = t & 15;
    if (e >= N_EDGES) return;
    int s = ei[e], d = ei[N_EDGES + e];
    float norm = dinv[s] * ew[e] * dinv[d];
    const float4 hv = *(const float4*)(h + (size_t)s * HIDDEN + j * 4);
    float* dp = agg + (size_t)d * HIDDEN + j * 4;
    atomicAdd(dp + 0, norm * hv.x);
    atomicAdd(dp + 1, norm * hv.y);
    atomicAdd(dp + 2, norm * hv.z);
    atomicAdd(dp + 3, norm * hv.w);
}

// layer 2: ew = 1; 10 threads per edge, 4 of 40 classes each
__global__ void gcn_scatter2(const int* __restrict__ ei, const float* __restrict__ dinv,
                             const float* __restrict__ h, float* __restrict__ agg) {
    int t = blockIdx.x * blockDim.x + threadIdx.x;
    int e = t / 10;
    int j = t - e * 10;
    if (e >= N_EDGES) return;
    int s = ei[e], d = ei[N_EDGES + e];
    float norm = dinv[s] * dinv[d];
    const float4 hv = *(const float4*)(h + (size_t)s * N_CLASSES + j * 4);
    float* dp = agg + (size_t)d * N_CLASSES + j * 4;
    atomicAdd(dp + 0, norm * hv.x);
    atomicAdd(dp + 1, norm * hv.y);
    atomicAdd(dp + 2, norm * hv.z);
    atomicAdd(dp + 3, norm * hv.w);
}

// out1 = relu(agg + dinv^2 * h + b)  (in place into agg)
__global__ void gcn_combine1(float* __restrict__ agg, const float* __restrict__ h,
                             const float* __restrict__ dinv, const float* __restrict__ b) {
    int t = blockIdx.x * blockDim.x + threadIdx.x;
    if (t >= N_NODES * HIDDEN) return;
    int i = t >> 6, f = t & 63;
    float di = dinv[i];
    float v = agg[t] + di * di * h[t] + b[f];
    agg[t] = fmaxf(v, 0.0f);
}

// final combine + log_softmax over 40 classes, 1 thread per node
__global__ void gcn_final(const float* __restrict__ agg, const float* __restrict__ h,
                          const float* __restrict__ dinv, const float* __restrict__ b,
                          float* __restrict__ out) {
    int i = blockIdx.x * blockDim.x + threadIdx.x;
    if (i >= N_NODES) return;
    float di = dinv[i], dd = di * di;
    float z[N_CLASSES];
    float m = -3.0e38f;
#pragma unroll
    for (int c = 0; c < N_CLASSES; ++c) {
        float v = agg[(size_t)i * N_CLASSES + c] + dd * h[(size_t)i * N_CLASSES + c] + b[c];
        z[c] = v;
        m = fmaxf(m, v);
    }
    float s = 0.0f;
#pragma unroll
    for (int c = 0; c < N_CLASSES; ++c) s += __expf(z[c] - m);
    float lse = m + __logf(s);
#pragma unroll
    for (int c = 0; c < N_CLASSES; ++c) out[(size_t)i * N_CLASSES + c] = z[c] - lse;
}

// ---------------------------------------------------------------- launcher
extern "C" void kernel_launch(void* const* d_in, const int* in_sizes, int n_in,
                              void* d_out, int out_size, void* d_ws, size_t ws_size,
                              hipStream_t stream) {
    (void)in_sizes; (void)n_in; (void)out_size; (void)ws_size;
    const float* x  = (const float*)d_in[0];
    const int*   ei = (const int*)  d_in[1];
    const float* ew = (const float*)d_in[2];
    const float* W1 = (const float*)d_in[3];
    const float* b1 = (const float*)d_in[4];
    const float* W2 = (const float*)d_in[5];
    const float* b2 = (const float*)d_in[6];
    float* out = (float*)d_out;
    float* ws  = (float*)d_ws;

    // workspace layout (float offsets, all >=32B aligned)
    float*  dinv1 = ws;                       //   100,000
    float*  dinv2 = ws + 100000;              //   100,000
    float*  h1    = ws + 200000;              // 6,400,000  (reused as h2)
    float*  agg1  = ws + 6600000;             // 6,400,000  (becomes out1 in place)
    float*  agg2  = ws + 13000000;            // 4,000,000
    __bf16* WT1   = (__bf16*)(ws + 17000000); // 16,384 bf16  (8,192 floats)
    __bf16* WT2   = (__bf16*)(ws + 17008192); //  3,072 bf16
    float*  h2    = h1;
    float*  out1  = agg1;

    const int B = 256;
    // zero degree buffers + both accumulators (agg1,agg2 contiguous)
    gcn_fill0<<<(200000 + B - 1) / B, B, 0, stream>>>(ws, 200000);
    gcn_fill0<<<(10400000 + B - 1) / B, B, 0, stream>>>(agg1, 10400000);

    // prep transposed bf16 weights
    gcn_prepW1<<<(HIDDEN * N_FEAT + B - 1) / B, B, 0, stream>>>(W1, WT1);
    gcn_prepW2<<<(NPAD2 * HIDDEN + B - 1) / B, B, 0, stream>>>(W2, WT2);

    gcn_degree<<<(N_EDGES + B - 1) / B, B, 0, stream>>>(ei, ew, dinv1, dinv2);
    gcn_rsqrt <<<(N_NODES + B - 1) / B, B, 0, stream>>>(dinv1, dinv2);

    const int gemmBlocks = (N_NODES / 16 + 3) / 4;   // 6250 waves, 4 waves/block
    gcn_gemm1<<<gemmBlocks, 128, 0, stream>>>(x, WT1, h1);

    gcn_scatter1<<<((size_t)N_EDGES * 16 + B - 1) / B, B, 0, stream>>>(ei, ew, dinv1, h1, agg1);
    gcn_combine1<<<(N_NODES * HIDDEN + B - 1) / B, B, 0, stream>>>(agg1, h1, dinv1, b1);

    gcn_gemm2<<<gemmBlocks, 128, 0, stream>>>(out1, WT2, h2);

    gcn_scatter2<<<((size_t)N_EDGES * 10 + B - 1) / B, B, 0, stream>>>(ei, dinv2, h2, agg2);
    gcn_final<<<(N_NODES + B - 1) / B, B, 0, stream>>>(agg2, h2, dinv2, b2, out);
}
